// MoELayer_16690242912310
// MI455X (gfx1250) — compile-verified, matrix operand order checked by reading, not run on hardware
//
#include <hip/hip_runtime.h>
#include <hip/hip_bf16.h>
#include <math.h>

// ---------------------------------------------------------------------------
// MoE (top-2 of 8 experts), S=2048 B=2 D=1024 F=4096, fp32 in/out.
//
// MI455X reasoning: routed compute = 137 GFLOP (dense = 4x for nothing).
// bf16 WMMA (16x16x32, f32 accum) is the right pipe; f32 WMMA is 8x slower.
// Round-1 lesson: building B fragments from fp32 global (16 b32 loads +
// 16 cvt per WMMA) makes the loop VALU/VMEM issue-bound.  So: pre-swizzle
// weights to bf16 in the exact per-lane WMMA fragment layout (16 contiguous
// bf16 per lane, 32B aligned) -> hot loop is 2 ds_load_b128 (A, reused over
// 2-4 tiles) + 2 global_load_b128 per B + 1 v_wmma.  Extra traffic for the
// one-shot convert (~540 MB total ~ 23us @ 23.3 TB/s) beats a 10x+ issue
// bottleneck.  L2 (192 MB) holds the 128 MB of bf16 weights.
// ---------------------------------------------------------------------------

#define TOKENS   4096   // S*B
#define D_DIM    1024
#define F_DIM    4096
#define NEXP     8
#define FCHUNK   512    // F columns per LDS-resident H chunk
#define TILE_M   16     // tokens per block
#define NTHREADS 512    // 16 waves

typedef __attribute__((ext_vector_type(16))) __bf16 v16bf;
typedef __attribute__((ext_vector_type(8)))  float  v8f;

__device__ __forceinline__ v8f wmma_bf16(v16bf a, v16bf b, v8f c) {
    // (neg_a, A, neg_b, B, c_mod, C, reuse_a, reuse_b)
    return __builtin_amdgcn_wmma_f32_16x16x32_bf16(false, a, false, b,
                                                   (short)0, c, false, false);
}

// ---------------------------------------------------------------------------
// Weight pre-swizzle: fp32 [E][K][N] row-major -> bf16 fragment groups.
// Group g = ((e*(N/16) + ntile)*(K/32) + kb)*32 + lane holds the 16 K-values
// (r = 0..15) for column n = ntile*16 + (lane&15), K-half (lane>>4):
//   value[r] = src[e][kb*32 + (lane>>4)*16 + r][n]
// Consumer lane loads its whole B fragment as one 32-byte vector.
// ---------------------------------------------------------------------------
__global__ __launch_bounds__(256)
void swizzle_kernel(const float* __restrict__ src, __bf16* __restrict__ dst,
                    int K, int N) {
    const size_t g   = (size_t)blockIdx.x * 256 + threadIdx.x;
    const int   lane = (int)(g & 31);
    const int   kbt  = K / 32, ntl = N / 16;
    size_t tmp = g >> 5;
    const int kb = (int)(tmp % kbt); tmp /= kbt;
    const int nt = (int)(tmp % ntl); tmp /= ntl;
    const int e  = (int)tmp;
    const int kbase = kb * 32 + (lane >> 4) * 16;
    const int n     = nt * 16 + (lane & 15);
    const float* s = src + ((size_t)e * K + kbase) * N + n;   // coalesced rows
    v16bf v;
#pragma unroll
    for (int r = 0; r < 16; ++r) v[r] = (__bf16)s[(size_t)r * N];
    *((v16bf*)dst + g) = v;                                   // coalesced 32B
}

// ---------------------------------------------------------------------------
// Kernel 1: gating.  One block (8 waves) per token; wave e computes logit e.
// ---------------------------------------------------------------------------
__global__ __launch_bounds__(256)
void gate_kernel(const float* __restrict__ x,
                 const float* __restrict__ gate_w,
                 const float* __restrict__ gate_b,
                 int*   __restrict__ counts,
                 int*   __restrict__ tok_idx,   // [NEXP][TOKENS]
                 float* __restrict__ tok_p) {   // [NEXP][TOKENS]
    const int t    = blockIdx.x;
    const int wave = threadIdx.x >> 5;   // expert id
    const int lane = threadIdx.x & 31;

    __shared__ float logits[NEXP];

    const float* xr = x + (size_t)t * D_DIM;
    float acc = 0.f;
    for (int d = lane; d < D_DIM; d += 32)
        acc += xr[d] * gate_w[d * NEXP + wave];
#pragma unroll
    for (int off = 16; off > 0; off >>= 1)
        acc += __shfl_down(acc, off, 32);
    if (lane == 0) logits[wave] = acc + gate_b[wave];
    __syncthreads();

    if (threadIdx.x == 0) {
        int i0 = 0;
        for (int e = 1; e < NEXP; ++e) if (logits[e] > logits[i0]) i0 = e;
        int i1 = (i0 == 0) ? 1 : 0;
        for (int e = 0; e < NEXP; ++e) {
            if (e == i0) continue;
            if (logits[e] > logits[i1]) i1 = e;
        }
        float m  = fmaxf(logits[i0], logits[i1]);
        float e0 = __expf(logits[i0] - m);
        float e1 = __expf(logits[i1] - m);
        float inv = 1.f / (e0 + e1);
        int s0 = atomicAdd(&counts[i0], 1);
        tok_idx[i0 * TOKENS + s0] = t;
        tok_p  [i0 * TOKENS + s0] = e0 * inv;
        int s1 = atomicAdd(&counts[i1], 1);
        tok_idx[i1 * TOKENS + s1] = t;
        tok_p  [i1 * TOKENS + s1] = e1 * inv;
    }
}

// ---------------------------------------------------------------------------
// Kernel 2 (fast path): routed FFN on pre-swizzled bf16 weights.
// One block per (expert, 16-token tile); 16 waves.
//   phase 1: Hchunk[16][512] = relu(X*W1+b1) -> LDS bf16 (2 f-tiles/wave)
//   phase 2: Y[16][1024] += Hchunk*W2        (4 persistent f32 tiles/wave)
//   finalize: out[tok] += p * (Y + b2) via global f32 atomics.
// ---------------------------------------------------------------------------
__global__ __launch_bounds__(NTHREADS, 1)
void moe_ffn_bf16_kernel(const float* __restrict__ x,
                         const __bf16* __restrict__ w1s,
                         const float*  __restrict__ b1,
                         const __bf16* __restrict__ w2s,
                         const float*  __restrict__ b2,
                         const int*    __restrict__ counts,
                         const int*    __restrict__ tok_idx,
                         const float*  __restrict__ tok_p,
                         float* __restrict__ out) {
    const int e    = blockIdx.x >> 8;
    const int tile = blockIdx.x & 255;
    const int cnt  = counts[e];
    if (tile * TILE_M >= cnt) return;

    __shared__ __bf16 Xs[TILE_M][D_DIM];    // 32 KB activation tile
    __shared__ __bf16 Hs[TILE_M][FCHUNK];   // 16 KB hidden chunk
    __shared__ int    tok_s[TILE_M];
    __shared__ float  p_s[TILE_M];

    const int tid   = threadIdx.x;
    const int wave  = tid >> 5;
    const int lane  = tid & 31;
    const int nIdx  = lane & 15;
    const int kHalf = lane >> 4;

    if (tid < TILE_M) {
        int  slot  = tile * TILE_M + tid;
        bool valid = slot < cnt;
        tok_s[tid] = valid ? tok_idx[e * TOKENS + slot] : 0;
        p_s[tid]   = valid ? tok_p[e * TOKENS + slot]   : 0.f;  // p=0 => +0.0
    }
    __syncthreads();

    for (int i = tid; i < TILE_M * D_DIM; i += NTHREADS) {
        int r = i >> 10, d = i & (D_DIM - 1);
        Xs[r][d] = (__bf16)x[(size_t)tok_s[r] * D_DIM + d];
    }
    __syncthreads();

    const v16bf* B1 = (const v16bf*)w1s;
    const v16bf* B2 = (const v16bf*)w2s;

    v8f yacc[4] = {};   // this wave's 16x64 slice of Y

    for (int c = 0; c < F_DIM / FCHUNK; ++c) {
        // -------- phase 1: 2 f-tiles per wave, A reused across tiles ------
        const int lt0 = wave * 2;                    // local tile 0..31
        v8f hacc[2] = {};
        for (int kb = 0; kb < D_DIM / 32; ++kb) {
            v16bf a = *(const v16bf*)&Xs[nIdx][kb * 32 + kHalf * 16];
            // fragment group base for (e, ft, kb): 32 lanes per group row
            size_t g0 = (((size_t)e * (F_DIM / 16) + (c * 32 + lt0))
                         * (D_DIM / 32) + kb) * 32 + lane;
            __builtin_prefetch(&B1[g0 + 32], 0, 1);  // next k-slab
            hacc[0] = wmma_bf16(a, B1[g0], hacc[0]);
            hacc[1] = wmma_bf16(a, B1[g0 + (size_t)(D_DIM / 32) * 32], hacc[1]);
        }
        __syncthreads();              // prior phase-2 reads of Hs are done
#pragma unroll
        for (int t = 0; t < 2; ++t) {
            const int   col   = (lt0 + t) * 16 + nIdx;       // within chunk
            const float bias1 = b1[e * F_DIM + c * FCHUNK + col];
#pragma unroll
            for (int v = 0; v < 8; ++v) {                    // M = v + 8*kHalf
                float h = hacc[t][v] + bias1;
                Hs[v + 8 * kHalf][col] = (__bf16)fmaxf(h, 0.f);
            }
        }
        __syncthreads();

        // -------- phase 2: 4 d-tiles per wave, A reused across tiles ------
        for (int kbl = 0; kbl < FCHUNK / 32; ++kbl) {
            const int kbf = c * (FCHUNK / 32) + kbl;
            v16bf a = *(const v16bf*)&Hs[nIdx][kbl * 32 + kHalf * 16];
            size_t g0 = (((size_t)e * (D_DIM / 16) + wave * 4)
                         * (F_DIM / 32) + kbf) * 32 + lane;
            __builtin_prefetch(&B2[g0 + 32], 0, 1);
#pragma unroll
            for (int t = 0; t < 4; ++t)
                yacc[t] = wmma_bf16(a, B2[g0 + (size_t)t * (F_DIM / 32) * 32],
                                    yacc[t]);
        }
    }

#pragma unroll
    for (int t = 0; t < 4; ++t) {
        const int   dcol  = (wave * 4 + t) * 16 + nIdx;
        const float bias2 = b2[e * D_DIM + dcol];
#pragma unroll
        for (int v = 0; v < 8; ++v) {
            int   M   = v + 8 * kHalf;
            float val = p_s[M] * (yacc[t][v] + bias2);
            atomicAdd(&out[(size_t)tok_s[M] * D_DIM + dcol], val);
        }
    }
}

// ---------------------------------------------------------------------------
// Kernel 2 (fallback, small workspace): fp32 weights read directly.
// ---------------------------------------------------------------------------
__device__ __forceinline__ v16bf load_b_frag(const float* __restrict__ base,
                                             int ld) {
    v16bf b;
#pragma unroll
    for (int r = 0; r < 16; ++r)
        b[r] = (__bf16)base[(size_t)r * ld];
    return b;
}

__global__ __launch_bounds__(NTHREADS, 1)
void moe_ffn_fp32_kernel(const float* __restrict__ x,
                         const float* __restrict__ w1, const float* __restrict__ b1,
                         const float* __restrict__ w2, const float* __restrict__ b2,
                         const int*   __restrict__ counts,
                         const int*   __restrict__ tok_idx,
                         const float* __restrict__ tok_p,
                         float* __restrict__ out) {
    const int e    = blockIdx.x >> 8;
    const int tile = blockIdx.x & 255;
    const int cnt  = counts[e];
    if (tile * TILE_M >= cnt) return;

    __shared__ __bf16 Xs[TILE_M][D_DIM];
    __shared__ __bf16 Hs[TILE_M][256];
    __shared__ int    tok_s[TILE_M];
    __shared__ float  p_s[TILE_M];

    const int tid   = threadIdx.x;
    const int wave  = tid >> 5;
    const int lane  = tid & 31;
    const int nIdx  = lane & 15;
    const int kHalf = lane >> 4;

    if (tid < TILE_M) {
        int  slot  = tile * TILE_M + tid;
        bool valid = slot < cnt;
        tok_s[tid] = valid ? tok_idx[e * TOKENS + slot] : 0;
        p_s[tid]   = valid ? tok_p[e * TOKENS + slot]   : 0.f;
    }
    __syncthreads();
    for (int i = tid; i < TILE_M * D_DIM; i += NTHREADS) {
        int r = i >> 10, d = i & (D_DIM - 1);
        Xs[r][d] = (__bf16)x[(size_t)tok_s[r] * D_DIM + d];
    }
    __syncthreads();

    const float* W1 = w1 + (size_t)e * D_DIM * F_DIM;
    const float* W2 = w2 + (size_t)e * F_DIM * D_DIM;
    v8f yacc[4] = {};

    for (int c = 0; c < F_DIM / 256; ++c) {
        const int fcol = c * 256 + wave * 16;
        v8f hacc = {};
        for (int kb = 0; kb < D_DIM / 32; ++kb) {
            const int krow = kb * 32 + kHalf * 16;
            v16bf a = *(const v16bf*)&Xs[nIdx][krow];
            v16bf b = load_b_frag(W1 + (size_t)krow * F_DIM + fcol + nIdx, F_DIM);
            hacc = wmma_bf16(a, b, hacc);
        }
        const float bias1 = b1[e * F_DIM + fcol + nIdx];
        __syncthreads();
#pragma unroll
        for (int v = 0; v < 8; ++v)
            Hs[v + 8 * kHalf][wave * 16 + nIdx] =
                (__bf16)fmaxf(hacc[v] + bias1, 0.f);
        __syncthreads();
#pragma unroll
        for (int t = 0; t < 4; ++t) {
            const int dcol = (wave * 4 + t) * 16 + nIdx;
#pragma unroll
            for (int kb = 0; kb < 256 / 32; ++kb) {
                const int kloc = kb * 32 + kHalf * 16;
                v16bf a = *(const v16bf*)&Hs[nIdx][kloc];
                v16bf b = load_b_frag(W2 + (size_t)(c * 256 + kloc) * D_DIM + dcol,
                                      D_DIM);
                yacc[t] = wmma_bf16(a, b, yacc[t]);
            }
        }
    }
#pragma unroll
    for (int t = 0; t < 4; ++t) {
        const int   dcol  = (wave * 4 + t) * 16 + nIdx;
        const float bias2 = b2[e * D_DIM + dcol];
#pragma unroll
        for (int v = 0; v < 8; ++v) {
            int   M   = v + 8 * kHalf;
            atomicAdd(&out[(size_t)tok_s[M] * D_DIM + dcol],
                      p_s[M] * (yacc[t][v] + bias2));
        }
    }
}

// ---------------------------------------------------------------------------
// Host launcher (graph-capture safe).
// Workspace (fast path): W1s 64MB | W2s 64MB | counts | tok_idx | tok_p
// ---------------------------------------------------------------------------
extern "C" void kernel_launch(void* const* d_in, const int* in_sizes, int n_in,
                              void* d_out, int out_size, void* d_ws, size_t ws_size,
                              hipStream_t stream) {
    (void)in_sizes; (void)n_in;
    const float* x      = (const float*)d_in[0];
    const float* gate_w = (const float*)d_in[1];
    const float* gate_b = (const float*)d_in[2];
    const float* w1     = (const float*)d_in[3];
    const float* b1     = (const float*)d_in[4];
    const float* w2     = (const float*)d_in[5];
    const float* b2     = (const float*)d_in[6];
    float* out = (float*)d_out;

    const size_t wbytes     = (size_t)NEXP * D_DIM * F_DIM * sizeof(__bf16); // 64MB
    const size_t route_need = (size_t)(NEXP + 2 * NEXP * TOKENS) * 4 + 64;
    const bool   fast       = ws_size >= 2 * wbytes + route_need;

    hipMemsetAsync(out, 0, (size_t)out_size * sizeof(float), stream);

    if (fast) {
        __bf16* w1s    = (__bf16*)d_ws;
        __bf16* w2s    = (__bf16*)((char*)d_ws + wbytes);
        int*    counts = (int*)((char*)d_ws + 2 * wbytes);
        int*    tok_i  = counts + NEXP;
        float*  tok_p  = (float*)(tok_i + NEXP * TOKENS);

        hipMemsetAsync(counts, 0, NEXP * sizeof(int), stream);

        const int g1 = (NEXP * (F_DIM / 16) * (D_DIM / 32) * 32) / 256;
        const int g2 = (NEXP * (D_DIM / 16) * (F_DIM / 32) * 32) / 256;
        swizzle_kernel<<<g1, 256, 0, stream>>>(w1, w1s, D_DIM, F_DIM);
        swizzle_kernel<<<g2, 256, 0, stream>>>(w2, w2s, F_DIM, D_DIM);
        gate_kernel<<<TOKENS, 256, 0, stream>>>(x, gate_w, gate_b,
                                                counts, tok_i, tok_p);
        moe_ffn_bf16_kernel<<<NEXP * (TOKENS / TILE_M), NTHREADS, 0, stream>>>(
            x, w1s, b1, w2s, b2, counts, tok_i, tok_p, out);
    } else {
        int*   counts = (int*)d_ws;
        int*   tok_i  = counts + NEXP;
        float* tok_p  = (float*)(tok_i + NEXP * TOKENS);
        hipMemsetAsync(counts, 0, NEXP * sizeof(int), stream);
        gate_kernel<<<TOKENS, 256, 0, stream>>>(x, gate_w, gate_b,
                                                counts, tok_i, tok_p);
        moe_ffn_fp32_kernel<<<NEXP * (TOKENS / TILE_M), NTHREADS, 0, stream>>>(
            x, w1, b1, w2, b2, counts, tok_i, tok_p, out);
    }
}
